// MoEViTBlock_85736137163326
// MI455X (gfx1250) — matrix operand order repaired and checked by reading, not compiled, not run on hardware
//
#include <hip/hip_runtime.h>

// ---------------- problem constants ----------------
constexpr int S_   = 512;
constexpr int B_   = 16;
constexpr int D_   = 768;
constexpr int H_   = 12;
constexpr int E_   = 8;
constexpr int HID_ = 3072;
constexpr int HD_  = 64;           // D_/H_
constexpr int T_   = S_ * B_;      // 8192 tokens

// TDM path: ROCm 7.2 (clang-22) 5-arg builtin lane only.
#if __has_builtin(__builtin_amdgcn_tensor_load_to_lds) && \
    !__has_include(<hip/amd_detail/amd_gfx1250_TDM.h>)
#define USE_TDM5 1
#else
#define USE_TDM5 0
#endif

// ---------------- WMMA types -----------------------
typedef __attribute__((ext_vector_type(16))) __bf16 v16bf;
typedef __attribute__((ext_vector_type(8)))  float  v8f;
typedef __attribute__((ext_vector_type(4)))  unsigned int u32x4;
typedef __attribute__((ext_vector_type(8)))  int i32x8;
typedef __attribute__((ext_vector_type(4)))  int i32x4;

static __device__ inline v8f wmma_bf16(v16bf a, v16bf b, v8f c) {
  return __builtin_amdgcn_wmma_f32_16x16x32_bf16(false, a, false, b, (short)0, c,
                                                 false, false);
}

// Build a 16xbf16 fragment from two 16-byte chunks (no per-element repack).
static __device__ inline v16bf frag_from2(const void* p0, const void* p1) {
  union { v16bf v; uint4 q[2]; } f;
  f.q[0] = *(const uint4*)p0;
  f.q[1] = *(const uint4*)p1;
  return f.v;
}

// A fragment (16M x 32K bf16) from a row-major tile with row stride ld elems.
static __device__ inline v16bf load_a_frag(const __bf16* base, int ld) {
  int lane = threadIdx.x & 31;
  int lh = lane >> 4, m = lane & 15;
  const __bf16* p = base + m * ld + lh * 8;
  return frag_from2(p, p + 16);
}

// B fragment (32K x 16N bf16) from a TRANSPOSED tile BsT[n][k].
static __device__ inline v16bf load_b_fragT(const __bf16* baseT, int ld) {
  int lane = threadIdx.x & 31;
  int lh = lane >> 4, n = lane & 15;
  const __bf16* p = baseT + n * ld + lh * 8;
  return frag_from2(p, p + 16);
}

// ---------------- fp32 -> bf16 convert ----------------
__global__ void cvt_bf16_kernel(const float* __restrict__ in,
                                __bf16* __restrict__ out, int n) {
  int i = blockIdx.x * 256 + threadIdx.x;
  if (i < n) out[i] = (__bf16)in[i];
}

// ---------------- LayerNorm (one block per token) ----------------
__global__ __launch_bounds__(256)
void ln_kernel(const float* __restrict__ x, const float* __restrict__ w,
               const float* __restrict__ b, __bf16* __restrict__ out,
               __bf16* __restrict__ xcopy) {
  int t = blockIdx.x;
  const float* row = x + (size_t)t * D_;
  __shared__ float r1[256], r2[256];
  float s = 0.f, s2 = 0.f;
  for (int i = threadIdx.x; i < D_; i += 256) {
    float v = row[i];
    s += v; s2 += v * v;
    if (xcopy) xcopy[(size_t)t * D_ + i] = (__bf16)v;
  }
  r1[threadIdx.x] = s; r2[threadIdx.x] = s2;
  __syncthreads();
  for (int st = 128; st > 0; st >>= 1) {
    if (threadIdx.x < st) {
      r1[threadIdx.x] += r1[threadIdx.x + st];
      r2[threadIdx.x] += r2[threadIdx.x + st];
    }
    __syncthreads();
  }
  float mean = r1[0] * (1.0f / D_);
  float var  = r2[0] * (1.0f / D_) - mean * mean;
  float rstd = rsqrtf(var + 1e-5f);
  for (int i = threadIdx.x; i < D_; i += 256) {
    float v = row[i];
    out[(size_t)t * D_ + i] = (__bf16)((v - mean) * rstd * w[i] + b[i]);
  }
}

// ------------ generic bf16 GEMM (128x128 block, 8 waves of 64x32) ------------
// MODE 0: outb = bf16((acc + bias) * ascale)
// MODE 1: outf = resid + acc + bias                       (wo projection)
// MODE 2: outb = bf16(gelu(acc + bias))                   (MoE up)
// MODE 3: outf = resid + gate[row] * (acc + bias)         (MoE down -> d_out)
#define BM 128
#define BN 128
#define BK 32
#define LDA_P (BK + 8)   // padded LDS stride: 40 elems = 80B (16B multiple)

#if USE_TDM5
// One TDM copy of the A tile (BM x BK bf16) into padded LDS (D# per ISA ch.8).
static __device__ inline void tdm_load_a_tile(const __bf16* gsrc, void* ldst,
                                              int Mtot, int K) {
  unsigned long long ga = (unsigned long long)(uintptr_t)gsrc;
  unsigned lds = (unsigned)(uintptr_t)ldst;
  u32x4 g0;
  g0[0] = 1u;                                           // count=1, user D#
  g0[1] = lds;                                          // lds_addr
  g0[2] = (unsigned)(ga & 0xFFFFFFFFu);                 // global_addr lo
  g0[3] = (unsigned)((ga >> 32) & 0x01FFFFFFu) | (2u << 30);  // hi | type=2
  unsigned w0 = (1u << 16)            // data_size = 2 bytes
              | (1u << 20)            // pad_enable
              | (3u << 22)            // pad_interval: 16 DWORDs (64B row)
              | (3u << 25);           // pad_amount: 4 DWORDs (16B) -> 80B rows
  unsigned w1 = ((unsigned)K & 0xFFFFu) << 16;
  unsigned w2 = ((unsigned)K >> 16) | (((unsigned)Mtot & 0xFFFFu) << 16);
  unsigned w3 = ((unsigned)Mtot >> 16) | ((unsigned)BK << 16);  // tile_dim0=32
  unsigned w4 = (unsigned)BM;                                   // tile_dim1=128
  unsigned w5 = (unsigned)K;                                    // dim0_stride
  i32x8 g1;
  g1[0] = (int)w0; g1[1] = (int)w1; g1[2] = (int)w2; g1[3] = (int)w3;
  g1[4] = (int)w4; g1[5] = (int)w5; g1[6] = 0; g1[7] = 0;
  i32x4 gz = {0, 0, 0, 0};
  __builtin_amdgcn_tensor_load_to_lds(g0, g1, gz, gz, 0);
  __builtin_amdgcn_s_wait_tensorcnt(0);
}
#endif

template <bool GROUPED, int MODE>
__global__ __launch_bounds__(256)
void gemm_bf16_kernel(const __bf16* __restrict__ A, const __bf16* __restrict__ W,
                      const float* __restrict__ bias,
                      __bf16* __restrict__ outb, float* __restrict__ outf,
                      const float* __restrict__ resid, const float* __restrict__ gate,
                      const int* __restrict__ perm, const int* __restrict__ cnt,
                      int M, int N, int K, int mtiles, float ascale) {
  int mt = blockIdx.x;
  const int* map = nullptr;
  const __bf16* Wp = W;
  const float* bp = bias;
  if constexpr (GROUPED) {
    int e = blockIdx.x / mtiles;
    mt    = blockIdx.x % mtiles;
    int rows = cnt[e];
    if (mt * BM >= rows) return;
    map = perm + (size_t)e * M;
    Wp  = W + (size_t)e * K * N;
    bp  = bias + (size_t)e * N;
  }
  int nb = blockIdx.y * BN;

  __shared__ __bf16 As[BM][LDA_P];       // row-major
  __shared__ __bf16 BsT[BN][LDA_P];      // transposed: BsT[n][k]

  int tid  = threadIdx.x;
  int wv   = tid >> 5;
  int lane = tid & 31;
  int lh   = lane >> 4;
  int ln16 = lane & 15;
  int wm   = wv & 1;        // M offset wm*64
  int wn   = wv >> 1;       // N offset wn*32

  // ---- staging coordinates (invariant over K loop) ----
  int arow_t0 = tid >> 2, arow_t1 = (tid + 256) >> 2;
  int ac8     = (tid & 3) * 8;
  int brow0 = tid >> 4,       brow1 = (tid + 256) >> 4;   // B K rows
  int bc8   = (tid & 15) * 8;                             // B N offset
  const __bf16* wptr0 = Wp + (size_t)brow0 * N + nb + bc8;
  const __bf16* wptr1 = Wp + (size_t)brow1 * N + nb + bc8;

  bool va0 = true, va1 = true;
  const __bf16* aptr0;
  const __bf16* aptr1;
  if constexpr (GROUPED) {
    int amap0 = map[mt * BM + arow_t0];
    int amap1 = map[mt * BM + arow_t1];
    va0 = amap0 >= 0;
    va1 = amap1 >= 0;
    aptr0 = A + (size_t)(va0 ? amap0 : 0) * K + ac8;
    aptr1 = A + (size_t)(va1 ? amap1 : 0) * K + ac8;
  } else {
    aptr0 = A + (size_t)(mt * BM + arow_t0) * K + ac8;
    aptr1 = A + (size_t)(mt * BM + arow_t1) * K + ac8;
  }

  v8f acc[4][2];
#pragma unroll
  for (int im = 0; im < 4; ++im)
#pragma unroll
    for (int in_ = 0; in_ < 2; ++in_) acc[im][in_] = (v8f){0,0,0,0,0,0,0,0};

  const int ktiles = K / BK;
  for (int kt = 0; kt < ktiles; ++kt) {
    // ---- stage A ----
#if USE_TDM5
    if constexpr (!GROUPED) {
      if (wv == 0)   // one wave issues the block's DMA; TDM ignores EXEC
        tdm_load_a_tile(A + (size_t)(mt * BM) * K + kt * BK, &As[0][0], M, K);
    } else
#endif
    {
      uint4 av0 = make_uint4(0u, 0u, 0u, 0u), av1 = av0;
      if (va0) av0 = *(const uint4*)aptr0;
      if (va1) av1 = *(const uint4*)aptr1;
      *(uint4*)(&As[arow_t0][ac8]) = av0;
      *(uint4*)(&As[arow_t1][ac8]) = av1;
    }
    // ---- stage B (transposed) ----
    uint4 bv0 = *(const uint4*)wptr0;
    uint4 bv1 = *(const uint4*)wptr1;
    if (kt + 1 < ktiles) {
      __builtin_prefetch(wptr0 + (size_t)BK * N, 0, 3);
      __builtin_prefetch(wptr1 + (size_t)BK * N, 0, 3);
    }
    {
      const __bf16* bvp0 = (const __bf16*)&bv0;
      const __bf16* bvp1 = (const __bf16*)&bv1;
#pragma unroll
      for (int q = 0; q < 8; ++q) BsT[bc8 + q][brow0] = bvp0[q];
#pragma unroll
      for (int q = 0; q < 8; ++q) BsT[bc8 + q][brow1] = bvp1[q];
    }
    aptr0 += BK; aptr1 += BK;
    wptr0 += (size_t)BK * N; wptr1 += (size_t)BK * N;
    __syncthreads();

    v16bf b0 = load_b_fragT(&BsT[wn * 32][0],      LDA_P);
    v16bf b1 = load_b_fragT(&BsT[wn * 32 + 16][0], LDA_P);
#pragma unroll
    for (int im = 0; im < 4; ++im) {
      v16bf a = load_a_frag(&As[wm * 64 + im * 16][0], LDA_P);
      acc[im][0] = wmma_bf16(a, b0, acc[im][0]);
      acc[im][1] = wmma_bf16(a, b1, acc[im][1]);
    }
    __syncthreads();
  }

  // ---- epilogue (single MODE compiled per instantiation) ----
#pragma unroll
  for (int im = 0; im < 4; ++im) {
#pragma unroll
    for (int r = 0; r < 8; ++r) {
      int lrow = wm * 64 + im * 16 + r + lh * 8;
      int slot = mt * BM + lrow;
      int trow = slot;
      if constexpr (GROUPED) {
        trow = map[slot];
        if (trow < 0) continue;
      }
#pragma unroll
      for (int in_ = 0; in_ < 2; ++in_) {
        float v = acc[im][in_][r];
        int col = nb + wn * 32 + in_ * 16 + ln16;
        v += bp[col];
        size_t oidx = (size_t)trow * N + col;
        if constexpr (MODE == 0) {
          outb[oidx] = (__bf16)(v * ascale);
        } else if constexpr (MODE == 1) {
          outf[oidx] = resid[oidx] + v;
        } else if constexpr (MODE == 2) {
          float g = 0.5f * v * (1.0f + erff(v * 0.70710678118654752f));
          outb[oidx] = (__bf16)g;
        } else {
          outf[oidx] = resid[oidx] + gate[trow] * v;
        }
      }
    }
  }
}

// ---------------- flash attention ----------------
// grid.x = B*H, grid.y = S/128 ; block = 256 (8 waves x 16 queries).
// Q is pre-scaled by 1/sqrt(HD) in the Q-projection epilogue.
__global__ __launch_bounds__(256)
void attn_kernel(const __bf16* __restrict__ q16, const __bf16* __restrict__ k16,
                 const __bf16* __restrict__ v16, __bf16* __restrict__ attn16) {
  int bh = blockIdx.x;
  int b  = bh / H_;
  int h  = bh % H_;
  int tid  = threadIdx.x;
  int wv   = tid >> 5;
  int lane = tid & 31;
  int lh   = lane >> 4;
  int ln16 = lane & 15;
  int qbase = blockIdx.y * 128 + wv * 16;

  __shared__ __bf16 Vs[HD_][40];       // transposed V chunk: Vs[dim][key]
  __shared__ float  Pw[8][16][40];     // per-wave raw f32 score bounce
  __shared__ float  Aux[8][16];        // per-wave per-row alpha / 1/l

  const __bf16* qrow =
      q16 + ((size_t)(qbase + ln16) * B_ + b) * D_ + h * HD_;
  v16bf aq0 = frag_from2(qrow + lh * 8,      qrow + 16 + lh * 8);
  v16bf aq1 = frag_from2(qrow + 32 + lh * 8, qrow + 48 + lh * 8);

  v8f z = {0,0,0,0,0,0,0,0};
  v8f o0 = z, o1 = z, o2 = z, o3 = z;
  float mstate = -1e30f, lstate = 0.f;   // per-lane state for row m = ln16

  int vkey = tid >> 3;
  int vd8  = (tid & 7) * 8;
  const __bf16* vptr = v16 + ((size_t)vkey * B_ + b) * D_ + h * HD_ + vd8;
  const size_t vstep = (size_t)32 * B_ * D_;

  for (int kc = 0; kc < S_; kc += 32) {
    // ---- cooperative transposed V staging: Vs[dim][key] ----
    {
      uint4 vvv = *(const uint4*)vptr;
      const __bf16* ve = (const __bf16*)&vvv;
#pragma unroll
      for (int q = 0; q < 8; ++q) Vs[vd8 + q][vkey] = ve[q];
      vptr += vstep;
    }
    __syncthreads();

    // ---- scores: two 16x16 key tiles; K^T frags packed from global ----
    v8f s0 = z, s1 = z;
#pragma unroll
    for (int kt = 0; kt < 2; ++kt) {
      const __bf16* krow =
          k16 + ((size_t)(kc + kt * 16 + ln16) * B_ + b) * D_ + h * HD_;
      v16bf bk0 = frag_from2(krow + lh * 8,      krow + 16 + lh * 8);
      v16bf bk1 = frag_from2(krow + 32 + lh * 8, krow + 48 + lh * 8);
      if (kt == 0) { s0 = wmma_bf16(aq0, bk0, s0); s0 = wmma_bf16(aq1, bk1, s0); }
      else         { s1 = wmma_bf16(aq0, bk0, s1); s1 = wmma_bf16(aq1, bk1, s1); }
    }

    // ---- bounce raw f32 scores C-layout -> A-layout ----
#pragma unroll
    for (int r = 0; r < 8; ++r) {
      int row = r + lh * 8;
      Pw[wv][row][ln16]      = s0[r];
      Pw[wv][row][16 + ln16] = s1[r];
    }
    union { float f[16]; uint4 q[4]; } eu;
    {
      const float* pr0 = &Pw[wv][ln16][lh * 8];        // cols lh*8 .. +7
      const float* pr1 = &Pw[wv][ln16][16 + lh * 8];   // cols 16+lh*8 .. +7
      eu.q[0] = *(const uint4*)pr0;
      eu.q[1] = *(const uint4*)(pr0 + 4);
      eu.q[2] = *(const uint4*)pr1;
      eu.q[3] = *(const uint4*)(pr1 + 4);
    }

    // ---- per-row softmax: 15 fmax + 1 shuffle; 16 exp + 15 add + 1 shuffle
    float lm = eu.f[0];
#pragma unroll
    for (int i = 1; i < 16; ++i) lm = fmaxf(lm, eu.f[i]);
    lm = fmaxf(lm, __shfl_xor(lm, 16, 32));
    float mnew  = fmaxf(mstate, lm);
    float alpha = __expf(mstate - mnew);
    mstate = mnew;

    v16bf pa;
    float rsum = 0.f;
#pragma unroll
    for (int i = 0; i < 16; ++i) {
      float p = __expf(eu.f[i] - mnew);
      rsum += p;
      pa[i] = (__bf16)p;
    }
    rsum += __shfl_xor(rsum, 16, 32);
    lstate = lstate * alpha + rsum;

    // ---- broadcast alpha to C-layout lanes via 16-float LDS array ----
    if (lh == 0) Aux[wv][ln16] = alpha;
    {
      union { float f[8]; uint4 q[2]; } au;
      const float* ar = &Aux[wv][lh * 8];
      au.q[0] = *(const uint4*)ar;
      au.q[1] = *(const uint4*)(ar + 4);
#pragma unroll
      for (int r = 0; r < 8; ++r) {
        o0[r] *= au.f[r]; o1[r] *= au.f[r];
        o2[r] *= au.f[r]; o3[r] *= au.f[r];
      }
    }

    // ---- P @ V over four 16-dim tiles ----
#pragma unroll
    for (int dt = 0; dt < 4; ++dt) {
      const __bf16* vb = &Vs[dt * 16 + ln16][lh * 8];
      v16bf bvv = frag_from2(vb, vb + 16);
      if      (dt == 0) o0 = wmma_bf16(pa, bvv, o0);
      else if (dt == 1) o1 = wmma_bf16(pa, bvv, o1);
      else if (dt == 2) o2 = wmma_bf16(pa, bvv, o2);
      else              o3 = wmma_bf16(pa, bvv, o3);
    }
    __syncthreads();   // protect Vs before next chunk overwrites it
  }

  // ---- normalize + store (broadcast 1/l the same way) ----
  if (lh == 0) Aux[wv][ln16] = 1.0f / lstate;
  union { float f[8]; uint4 q[2]; } iu;
  {
    const float* ar = &Aux[wv][lh * 8];
    iu.q[0] = *(const uint4*)ar;
    iu.q[1] = *(const uint4*)(ar + 4);
  }
#pragma unroll
  for (int r = 0; r < 8; ++r) {
    float inv = iu.f[r];
    int s = qbase + r + lh * 8;
    size_t base = ((size_t)s * B_ + b) * D_ + h * HD_;
    attn16[base +  0 + ln16] = (__bf16)(o0[r] * inv);
    attn16[base + 16 + ln16] = (__bf16)(o1[r] * inv);
    attn16[base + 32 + ln16] = (__bf16)(o2[r] * inv);
    attn16[base + 48 + ln16] = (__bf16)(o3[r] * inv);
  }
}

// ---------------- MoE routing ----------------
__global__ void init_route_kernel(int* __restrict__ cnt, int* __restrict__ perm) {
  int i = blockIdx.x * 256 + threadIdx.x;
  if (i < E_ * T_) perm[i] = -1;
  if (i < E_) cnt[i] = 0;
}

__global__ __launch_bounds__(256)
void route_kernel(const __bf16* __restrict__ t16, const float* __restrict__ wg,
                  const float* __restrict__ bg, float* __restrict__ gate,
                  int* __restrict__ eidx, int* __restrict__ cnt,
                  int* __restrict__ perm) {
  int wv   = threadIdx.x >> 5;
  int lane = threadIdx.x & 31;
  int t = blockIdx.x * 8 + wv;
  float acc[E_];
#pragma unroll
  for (int e = 0; e < E_; ++e) acc[e] = 0.f;
  for (int i = lane; i < D_; i += 32) {
    float tv = (float)t16[(size_t)t * D_ + i];
    const float* wr = wg + (size_t)i * E_;
#pragma unroll
    for (int e = 0; e < E_; ++e) acc[e] += tv * wr[e];
  }
#pragma unroll
  for (int e = 0; e < E_; ++e) {
#pragma unroll
    for (int msk = 16; msk > 0; msk >>= 1)
      acc[e] += __shfl_xor(acc[e], msk, 32);
    acc[e] += bg[e];
  }
  float mx = acc[0];
  int idx = 0;
#pragma unroll
  for (int e = 1; e < E_; ++e)
    if (acc[e] > mx) { mx = acc[e]; idx = e; }   // first max, matches argmax
  float se = 0.f;
#pragma unroll
  for (int e = 0; e < E_; ++e) se += __expf(acc[e] - mx);
  if (lane == 0) {
    gate[t] = 1.0f / se;
    eidx[t] = idx;
    int slot = atomicAdd(&cnt[idx], 1);
    perm[idx * T_ + slot] = t;
  }
}

// ---------------- host side ----------------
extern "C" void kernel_launch(void* const* d_in, const int* in_sizes, int n_in,
                              void* d_out, int out_size, void* d_ws, size_t ws_size,
                              hipStream_t stream) {
  (void)in_sizes; (void)n_in; (void)out_size; (void)ws_size;
  const float* x     = (const float*)d_in[0];
  const float* ln1_w = (const float*)d_in[1];
  const float* ln1_b = (const float*)d_in[2];
  const float* wq    = (const float*)d_in[3];
  const float* bq    = (const float*)d_in[4];
  const float* wk    = (const float*)d_in[5];
  const float* bk    = (const float*)d_in[6];
  const float* wvw   = (const float*)d_in[7];
  const float* bv    = (const float*)d_in[8];
  const float* wo    = (const float*)d_in[9];
  const float* bo    = (const float*)d_in[10];
  const float* ln2_w = (const float*)d_in[11];
  const float* ln2_b = (const float*)d_in[12];
  const float* wg    = (const float*)d_in[13];
  const float* bg    = (const float*)d_in[14];
  const float* w1    = (const float*)d_in[15];
  const float* b1    = (const float*)d_in[16];
  const float* w2    = (const float*)d_in[17];
  const float* b2    = (const float*)d_in[18];
  float* out = (float*)d_out;

  char* ws = (char*)d_ws;
  size_t off = 0;
  auto al = [&](size_t bytes) {
    size_t o = off;
    off = (off + bytes + 255) & ~(size_t)255;
    return o;
  };
  const size_t TD = (size_t)T_ * D_;
  __bf16* x16    = (__bf16*)(ws + al(TD * 2));
  __bf16* qin16  = (__bf16*)(ws + al(TD * 2));
  __bf16* q16    = (__bf16*)(ws + al(TD * 2));
  __bf16* k16    = (__bf16*)(ws + al(TD * 2));
  __bf16* v16    = (__bf16*)(ws + al(TD * 2));
  __bf16* attn16 = (__bf16*)(ws + al(TD * 2));
  float*  x1     = (float*) (ws + al(TD * 4));
  __bf16* t16    = (__bf16*)(ws + al(TD * 2));
  __bf16* h16    = (__bf16*)(ws + al((size_t)T_ * HID_ * 2));
  __bf16* wq16   = (__bf16*)(ws + al((size_t)D_ * D_ * 2));
  __bf16* wk16   = (__bf16*)(ws + al((size_t)D_ * D_ * 2));
  __bf16* wv16   = (__bf16*)(ws + al((size_t)D_ * D_ * 2));
  __bf16* wo16   = (__bf16*)(ws + al((size_t)D_ * D_ * 2));
  __bf16* w116   = (__bf16*)(ws + al((size_t)E_ * D_ * HID_ * 2));
  __bf16* w216   = (__bf16*)(ws + al((size_t)E_ * HID_ * D_ * 2));
  float*  gate   = (float*) (ws + al((size_t)T_ * 4));
  int*    eidx   = (int*)   (ws + al((size_t)T_ * 4));
  int*    cnt    = (int*)   (ws + al((size_t)E_ * 4));
  int*    perm   = (int*)   (ws + al((size_t)E_ * T_ * 4));

  dim3 blk(256);
  auto cvt = [&](const float* src, __bf16* dst, size_t n) {
    cvt_bf16_kernel<<<dim3((unsigned)((n + 255) / 256)), blk, 0, stream>>>(src, dst, (int)n);
  };
  cvt(wq, wq16, (size_t)D_ * D_);
  cvt(wk, wk16, (size_t)D_ * D_);
  cvt(wvw, wv16, (size_t)D_ * D_);
  cvt(wo, wo16, (size_t)D_ * D_);
  cvt(w1, w116, (size_t)E_ * D_ * HID_);
  cvt(w2, w216, (size_t)E_ * HID_ * D_);

  ln_kernel<<<dim3(T_), blk, 0, stream>>>(x, ln1_w, ln1_b, qin16, x16);

  // QKV projections; Q pre-scaled by 1/sqrt(HD)
  dim3 gproj(T_ / BM, D_ / BN);
  gemm_bf16_kernel<false, 0><<<gproj, blk, 0, stream>>>(qin16, wq16, bq, q16,
      nullptr, nullptr, nullptr, nullptr, nullptr, T_, D_, D_, 0, 0.125f);
  gemm_bf16_kernel<false, 0><<<gproj, blk, 0, stream>>>(x16, wk16, bk, k16,
      nullptr, nullptr, nullptr, nullptr, nullptr, T_, D_, D_, 0, 1.0f);
  gemm_bf16_kernel<false, 0><<<gproj, blk, 0, stream>>>(x16, wv16, bv, v16,
      nullptr, nullptr, nullptr, nullptr, nullptr, T_, D_, D_, 0, 1.0f);

  attn_kernel<<<dim3(B_ * H_, S_ / 128), blk, 0, stream>>>(q16, k16, v16, attn16);

  gemm_bf16_kernel<false, 1><<<gproj, blk, 0, stream>>>(attn16, wo16, bo, nullptr,
      x1, x, nullptr, nullptr, nullptr, T_, D_, D_, 0, 1.0f);

  ln_kernel<<<dim3(T_), blk, 0, stream>>>(x1, ln2_w, ln2_b, t16, nullptr);

  init_route_kernel<<<dim3((E_ * T_ + 255) / 256), blk, 0, stream>>>(cnt, perm);
  route_kernel<<<dim3(T_ / 8), blk, 0, stream>>>(t16, wg, bg, gate, eidx, cnt, perm);

  const int mtiles = T_ / BM;  // 64
  gemm_bf16_kernel<true, 2><<<dim3(E_ * mtiles, HID_ / BN), blk, 0, stream>>>(
      t16, w116, b1, h16, nullptr, nullptr, nullptr, perm, cnt,
      T_, HID_, D_, mtiles, 1.0f);

  gemm_bf16_kernel<true, 3><<<dim3(E_ * mtiles, D_ / BN), blk, 0, stream>>>(
      h16, w216, b2, nullptr, out, x1, gate, perm, cnt,
      T_, D_, HID_, mtiles, 1.0f);
}
